// SimplifiedSelfAttention_14766097564191
// MI455X (gfx1250) — matrix operand order repaired and checked by reading, not compiled, not run on hardware
//
#include <hip/hip_runtime.h>
#include <hip/hip_bf16.h>

// MI455X / gfx1250 flash-attention implementation of SimplifiedSelfAttention.
// wave32, WMMA 16x16x32 bf16 (f32 accum), TDM async tensor loads for K/V
// tile staging (double-buffered in LDS, TENSORcnt-tracked).
//
// Shapes: B=8, C=256, Cq=32, N=H*W=4096.
// Pipeline:
//   pack_weights : wq|wk|wv -> Wbf[320][256] bf16, Bias[320] f32
//   qkv_proj     : Y[320xN] = Wbf x X  (WMMA), writes
//                    Qbf[B][N][32]  (row = pixel, contiguous channels)
//                    Kbf[B][N][32]
//                    Vt [B][C][N]   (row = channel, contiguous m)
//   flash_attn   : block = 4 waves sharing one batch; K/V tiles staged via
//                  tensor_load_to_lds (TDM), each wave owns 16 query rows,
//                  online softmax over m in steps of 32, O[16x256] f32 in
//                  VGPRs; final (1+gamma)/l scale and floor(x*256)/256.

typedef __attribute__((ext_vector_type(16))) __bf16 v16bf;
typedef __attribute__((ext_vector_type(8)))  float  v8f;
typedef __attribute__((ext_vector_type(4)))  unsigned int v4u;
typedef __attribute__((ext_vector_type(8)))  int v8i;
typedef __attribute__((ext_vector_type(4)))  int v4i;

#define B_   8
#define C_   256
#define CQ_  32
#define N_   4096

__device__ __forceinline__ unsigned short f32_to_bf16(float f) {
    unsigned int u = __float_as_uint(f);
    unsigned int r = (u + 0x7FFFu + ((u >> 16) & 1u)) >> 16;
    return (unsigned short)r;
}

// 32 bytes of a fragment: v16bf view + two uint4 chunks for wide loads.
union FragU {
    v16bf v;
    uint4 q[2];
};

// ---------------------------------------------------------------------------
// TDM: async 2D tile load global -> LDS. D# encoding per CDNA5 ISA ch.8:
//   group0: [1:0]=count=1, [63:32]=lds_addr, [120:64]=global_addr,
//           [127:126]=type=2
//   group1: [17:16]=data_size=1 (2B), [79:48]=tensor_dim0, [111:80]=
//           tensor_dim1, [127:112]=tile_dim0, [143:128]=tile_dim1,
//           [207:160]=tensor_dim0_stride   (all in data_size units)
// Groups 2/3 zero (<=2D tensor). Tracked with TENSORcnt.
// ---------------------------------------------------------------------------
__device__ __forceinline__ void tdm_load_tile_2d(
    void* lds_dst, const void* gsrc,
    unsigned tensor_d0, unsigned tensor_d1,
    unsigned tile_d0, unsigned tile_d1,
    unsigned long long d0_stride)
{
    unsigned long long ga = (unsigned long long)(size_t)gsrc;
    unsigned lds_off = (unsigned)(size_t)lds_dst;   // low 32b = LDS offset

    v4u g0;
    g0[0] = 1u;                                      // count=1 (valid), user
    g0[1] = lds_off;
    g0[2] = (unsigned)(ga & 0xFFFFFFFFu);
    g0[3] = (unsigned)((ga >> 32) & 0x01FFFFFFu) | (2u << 30);  // type=2

    v8i g1;
    g1[0] = (int)(1u << 16);                         // data_size = 2 bytes
    g1[1] = (int)((tensor_d0 & 0xFFFFu) << 16);
    g1[2] = (int)((tensor_d0 >> 16) | ((tensor_d1 & 0xFFFFu) << 16));
    g1[3] = (int)((tensor_d1 >> 16) | ((tile_d0 & 0xFFFFu) << 16));
    g1[4] = (int)(tile_d1 & 0xFFFFu);                // tile_dim2 = 0
    g1[5] = (int)(d0_stride & 0xFFFFFFFFull);
    g1[6] = (int)((d0_stride >> 32) & 0xFFFFull);    // dim1 stride = 0
    g1[7] = 0;

    v4i z4 = {0, 0, 0, 0};
#if __clang_major__ >= 23
    v8i z8 = {0, 0, 0, 0, 0, 0, 0, 0};
    __builtin_amdgcn_tensor_load_to_lds(g0, g1, z4, z4, z8, 0);
#else
    __builtin_amdgcn_tensor_load_to_lds(g0, g1, z4, z4, 0);
#endif
}

// ---------------------------------------------------------------------------
// Kernel 0: pack weights to bf16 [320 x 256] and biases to f32 [320].
// Rows 0-31 = wq, 32-63 = wk, 64-319 = wv.
// ---------------------------------------------------------------------------
__global__ __launch_bounds__(256) void pack_weights(
    const float* __restrict__ wq, const float* __restrict__ bq,
    const float* __restrict__ wk, const float* __restrict__ bk,
    const float* __restrict__ wv, const float* __restrict__ bv,
    unsigned short* __restrict__ Wbf, float* __restrict__ Bias)
{
    int idx = blockIdx.x * 256 + threadIdx.x;
    if (idx < 320 * 256) {
        int o = idx >> 8, c = idx & 255;
        float w;
        if (o < 32)      w = wq[o * 256 + c];
        else if (o < 64) w = wk[(o - 32) * 256 + c];
        else             w = wv[(o - 64) * 256 + c];
        Wbf[idx] = f32_to_bf16(w);
    }
    if (idx < 320) {
        float bb;
        if (idx < 32)      bb = bq[idx];
        else if (idx < 64) bb = bk[idx - 32];
        else               bb = bv[idx - 64];
        Bias[idx] = bb;
    }
}

// ---------------------------------------------------------------------------
// Kernel 1: QKV projection. Block = 256 threads (8 waves), one batch and a
// 64-pixel tile. X tile (256ch x 64px) staged in LDS as bf16 (row = pixel,
// stride 264 elems: 16B-aligned rows). 20 M-tiles x 4 N-tiles = 80 wave
// jobs, each = 8 x wmma_f32_16x16x32_bf16 over K=256.
// ---------------------------------------------------------------------------
#define XS_STRIDE 264

__global__ __launch_bounds__(256) void qkv_proj(
    const float* __restrict__ x,
    const unsigned short* __restrict__ Wbf,
    const float* __restrict__ Bias,
    unsigned short* __restrict__ Qbf,
    unsigned short* __restrict__ Kbf,
    unsigned short* __restrict__ Vt)
{
    __shared__ __align__(16) unsigned short Xs[64 * XS_STRIDE];

    int b   = blockIdx.x >> 6;
    int n0  = (blockIdx.x & 63) * 64;
    int tid = threadIdx.x;

    const float* xb = x + (size_t)b * C_ * N_ + n0;
    for (int i = tid; i < 64 * 256; i += 256) {
        int c = i >> 6, p = i & 63;
        Xs[p * XS_STRIDE + c] = f32_to_bf16(xb[(size_t)c * N_ + p]);
    }
    __syncthreads();

    int lane = tid & 31;
    int wave = tid >> 5;
    int lcol = lane & 15;   // column-in-tile index
    int h    = lane >> 4;   // half-wave select

    v8f zero = {};

    for (int job = wave; job < 80; job += 8) {
        int mt = job >> 2;  // out-channel tile 0..19
        int nt = job & 3;   // pixel tile 0..3

        const uint4* wrow = (const uint4*)(Wbf + (size_t)(mt * 16 + lcol) * 256);
        const uint4* xcol = (const uint4*)(Xs + (nt * 16 + lcol) * XS_STRIDE);

        v8f acc = zero;
        #pragma unroll
        for (int kk = 0; kk < 8; ++kk) {
            FragU a, bf;
            // A 16x32: lanes 0-15 K{0-7,16-23}; lanes 16-31 K{8-15,24-31}
            a.q[0]  = wrow[kk * 4 + h];
            a.q[1]  = wrow[kk * 4 + h + 2];
            // B 32x16: lanes 0-15 K{0-7,16-23}; lanes 16-31 K{8-15,24-31}
            bf.q[0] = xcol[kk * 4 + h];
            bf.q[1] = xcol[kk * 4 + h + 2];
            acc = __builtin_amdgcn_wmma_f32_16x16x32_bf16(
                false, a.v, false, bf.v, (short)0, acc, false, false);
        }

        // C layout: VGPR g -> row M = g + 8h (out-channel), lane -> pixel.
        int nb = n0 + nt * 16 + lcol;
        #pragma unroll
        for (int g = 0; g < 8; ++g) {
            int o = mt * 16 + g + 8 * h;
            unsigned short bv16 = f32_to_bf16(acc[g] + Bias[o]);
            if (o < 32) {
                Qbf[((size_t)b * N_ + nb) * CQ_ + o] = bv16;
            } else if (o < 64) {
                Kbf[((size_t)b * N_ + nb) * CQ_ + (o - 32)] = bv16;
            } else {
                Vt[((size_t)b * C_ + (o - 64)) * N_ + nb] = bv16;
            }
        }
    }
}

// ---------------------------------------------------------------------------
// Kernel 2: flash attention. Block = 128 threads (4 waves), all on one
// batch, covering 64 query rows (wave w -> rows n0..n0+15). Per m-step of
// 32 keys:
//   - wave 0 issues TDM loads of next K tile (32x32, contiguous 2KB) and
//     next V tile (256 x 32, row stride N) into the spare LDS buffer, then
//     waits TENSORcnt for the current buffer; barrier releases all waves.
//   - S(16x32) = Q x K^T               -> 2 WMMA (B operands from LDS)
//   - online softmax (rows spread over 16 lanes; shfl_xor reductions)
//   - P: C-layout -> A-layout transpose through per-wave LDS staging
//   - O(16x256) += P x V'              -> 16 WMMA (B operands from LDS)
// ---------------------------------------------------------------------------
__global__ __launch_bounds__(128) void flash_attn(
    const unsigned short* __restrict__ Qbf,
    const unsigned short* __restrict__ Kbf,
    const unsigned short* __restrict__ Vt,
    const float* __restrict__ gamma,
    float* __restrict__ out)
{
    __shared__ __align__(16) unsigned short Ks[2][32 * 32];    //  4 KB
    __shared__ __align__(16) unsigned short Vs[2][32 * 256];   // 32 KB
    __shared__ __align__(16) unsigned short Pl[4 * 16 * 32];   //  4 KB

    int lane = threadIdx.x & 31;
    int wave = threadIdx.x >> 5;
    int b    = blockIdx.x >> 6;                       // 64 blocks per batch
    int n0   = ((blockIdx.x & 63) * 4 + wave) * 16;   // this wave's rows
    int lcol = lane & 15;
    int h    = lane >> 4;

    const unsigned short* Kb = Kbf + (size_t)b * N_ * CQ_;
    const unsigned short* Vb = Vt  + (size_t)b * C_ * N_;

    // Q fragment (A 16x32): row = pixel n0+lcol, 32 contiguous channels.
    FragU qf;
    {
        const uint4* qrow = (const uint4*)(Qbf + ((size_t)b * N_ + n0 + lcol) * CQ_);
        qf.q[0] = qrow[2 * h];
        qf.q[1] = qrow[2 * h + 1];
    }

    v8f zero = {};
    v8f acc[16];
    #pragma unroll
    for (int t = 0; t < 16; ++t) acc[t] = zero;

    float mi[8], li[8];
    #pragma unroll
    for (int g = 0; g < 8; ++g) { mi[g] = -INFINITY; li[g] = 0.0f; }

    unsigned short* Pw = Pl + wave * (16 * 32);

    // Prologue: stage tiles for m0 = 0 into buffer 0.
    if (wave == 0) {
        tdm_load_tile_2d(&Ks[0][0], Kb,           N_ * CQ_, 1,  1024, 1,   1024);
        tdm_load_tile_2d(&Vs[0][0], Vb,           N_,       C_, 32,   C_,  N_);
    }

    for (int it = 0; it < N_ / 32; ++it) {
        int m0  = it * 32;
        int cur = it & 1;
        if (wave == 0) {
            if (it + 1 < N_ / 32) {
                int nxt = cur ^ 1, m1 = m0 + 32;
                tdm_load_tile_2d(&Ks[nxt][0], Kb + (size_t)m1 * CQ_,
                                 N_ * CQ_, 1,  1024, 1,  1024);
                tdm_load_tile_2d(&Vs[nxt][0], Vb + m1,
                                 N_,       C_, 32,   C_, N_);
                __builtin_amdgcn_s_wait_tensorcnt((short)2);  // cur buf ready
            } else {
                __builtin_amdgcn_s_wait_tensorcnt((short)0);
            }
        }
        __syncthreads();   // current K/V tiles visible to all waves

        // K^T fragments (B 32x16) from LDS: column = key pixel m-local.
        FragU k0f, k1f;
        {
            const uint4* kr0 = (const uint4*)(&Ks[cur][0] + lcol * 32);
            const uint4* kr1 = (const uint4*)(&Ks[cur][0] + (16 + lcol) * 32);
            k0f.q[0] = kr0[2 * h]; k0f.q[1] = kr0[2 * h + 1];
            k1f.q[0] = kr1[2 * h]; k1f.q[1] = kr1[2 * h + 1];
        }
        v8f s0 = __builtin_amdgcn_wmma_f32_16x16x32_bf16(
            false, qf.v, false, k0f.v, (short)0, zero, false, false);
        v8f s1 = __builtin_amdgcn_wmma_f32_16x16x32_bf16(
            false, qf.v, false, k1f.v, (short)0, zero, false, false);

        // Online softmax. C layout: VGPR g -> row (g + 8h), lane -> column.
        float scl[8], p0[8], p1[8];
        #pragma unroll
        for (int g = 0; g < 8; ++g) {
            float a = s0[g], c = s1[g];
            float tv = fmaxf(a, c);
            #pragma unroll
            for (int off = 8; off >= 1; off >>= 1)
                tv = fmaxf(tv, __shfl_xor(tv, off, 32));
            float nm = fmaxf(mi[g], tv);
            float sc = __expf(mi[g] - nm);
            float e0 = __expf(a - nm);
            float e1 = __expf(c - nm);
            float rs = e0 + e1;
            #pragma unroll
            for (int off = 8; off >= 1; off >>= 1)
                rs += __shfl_xor(rs, off, 32);
            li[g] = li[g] * sc + rs;
            mi[g] = nm;
            scl[g] = sc; p0[g] = e0; p1[g] = e1;
        }
        #pragma unroll
        for (int t = 0; t < 16; ++t)
            #pragma unroll
            for (int g = 0; g < 8; ++g)
                acc[t][g] *= scl[g];

        // Transpose P (C layout) to A layout through per-wave LDS.
        #pragma unroll
        for (int g = 0; g < 8; ++g) {
            int r = g + 8 * h;
            Pw[r * 32 + lcol]      = f32_to_bf16(p0[g]);
            Pw[r * 32 + 16 + lcol] = f32_to_bf16(p1[g]);
        }
        __syncthreads();
        FragU pf;
        {
            const uint4* pr = (const uint4*)(Pw + lcol * 32);
            pf.q[0] = pr[2 * h];
            pf.q[1] = pr[2 * h + 1];
        }

        // O += P x V'.  V' fragment from LDS: column = channel, K = m.
        #pragma unroll
        for (int ct = 0; ct < 16; ++ct) {
            FragU vf;
            const uint4* vc = (const uint4*)(&Vs[cur][0] + (ct * 16 + lcol) * 32 + 16 * h);
            vf.q[0] = vc[0];
            vf.q[1] = vc[1];
            acc[ct] = __builtin_amdgcn_wmma_f32_16x16x32_bf16(
                false, pf.v, false, vf.v, (short)0, acc[ct], false, false);
        }
        __syncthreads();   // everyone done with buf[cur] before TDM reuses it
    }

    // Finalize: (1+gamma) * O / l, then floor(x*256)/256. VGPR g -> row
    // n0 + g + 8h, lane -> channel ct*16+lcol. Per lane: 8 consecutive n.
    float fac = 1.0f + gamma[0];
    float inv[8];
    #pragma unroll
    for (int g = 0; g < 8; ++g) inv[g] = fac / li[g];

    #pragma unroll
    for (int ct = 0; ct < 16; ++ct) {
        int c = ct * 16 + lcol;
        float* op = out + ((size_t)b * C_ + c) * N_ + n0 + 8 * h;
        float r[8];
        #pragma unroll
        for (int g = 0; g < 8; ++g)
            r[g] = floorf(acc[ct][g] * inv[g] * 256.0f) * 0.00390625f;
        float4 w0 = make_float4(r[0], r[1], r[2], r[3]);
        float4 w1 = make_float4(r[4], r[5], r[6], r[7]);
        *(float4*)(op)     = w0;
        *(float4*)(op + 4) = w1;
    }
}

// ---------------------------------------------------------------------------
// Launcher. Workspace layout (bytes):
//   [0, 163840)           Wbf  320x256 bf16
//   [163840, 165120)      Bias 320 f32          (pad to 165376)
//   [165376, +2097152)    Qbf  8x4096x32 bf16
//   [.., +2097152)        Kbf
//   [.., +16777216)       Vt   8x256x4096 bf16
// Total ~21.1 MB.
// ---------------------------------------------------------------------------
extern "C" void kernel_launch(void* const* d_in, const int* in_sizes, int n_in,
                              void* d_out, int out_size, void* d_ws, size_t ws_size,
                              hipStream_t stream) {
    (void)in_sizes; (void)n_in; (void)out_size; (void)ws_size;
    const float* x     = (const float*)d_in[0];
    const float* wq    = (const float*)d_in[1];
    const float* bq    = (const float*)d_in[2];
    const float* wk    = (const float*)d_in[3];
    const float* bk    = (const float*)d_in[4];
    const float* wv    = (const float*)d_in[5];
    const float* bv    = (const float*)d_in[6];
    const float* gamma = (const float*)d_in[7];
    float* out = (float*)d_out;

    char* ws = (char*)d_ws;
    unsigned short* Wbf  = (unsigned short*)(ws);
    float*          Bias = (float*)(ws + 163840);
    unsigned short* Qbf  = (unsigned short*)(ws + 165376);
    unsigned short* Kbf  = (unsigned short*)(ws + 165376 + 2097152);
    unsigned short* Vt   = (unsigned short*)(ws + 165376 + 2 * 2097152);

    pack_weights<<<(320 * 256 + 255) / 256, 256, 0, stream>>>(
        wq, bq, wk, bk, wv, bv, Wbf, Bias);
    qkv_proj<<<B_ * (N_ / 64), 256, 0, stream>>>(x, Wbf, Bias, Qbf, Kbf, Vt);
    flash_attn<<<B_ * (N_ / 64), 128, 0, stream>>>(Qbf, Kbf, Vt, gamma, out);
}
